// SparseGATConv_57561151701649
// MI455X (gfx1250) — compile-verified, matrix operand order checked by reading, not run on hardware
//
#include <hip/hip_runtime.h>
#include <hip/hip_bf16.h>

typedef __attribute__((ext_vector_type(16))) _Float16 v16h;
typedef __attribute__((ext_vector_type(8)))  _Float16 v8h;
typedef __attribute__((ext_vector_type(8)))  float    v8f;

#define N_NODES 50000
#define N_EDGES 800000
#define IN_F    64
#define OUT_F   64
#define HEADS   4
#define HF      (HEADS * OUT_F)   // 256
#define ALPHA   0.2f

// ---------------------------------------------------------------------------
// Init: zero out[] and denom[], fill e_max[] with -inf.
// ---------------------------------------------------------------------------
__global__ void gat_init(float* __restrict__ out, float* __restrict__ e_max,
                         float* __restrict__ denom) {
    const long long stride = (long long)gridDim.x * blockDim.x;
    long long i = (long long)blockIdx.x * blockDim.x + threadIdx.x;
    for (long long idx = i; idx < (long long)N_NODES * HF; idx += stride)
        out[idx] = 0.0f;
    for (long long idx = i; idx < (long long)N_NODES * HEADS; idx += stride) {
        e_max[idx] = -__builtin_huge_valf();
        denom[idx] = 0.0f;
    }
}

// ---------------------------------------------------------------------------
// Convert x -> f16 and W -> f16 *transposed* per head: Wt[head][o][f].
// Transposing W makes the WMMA B-fragment gather contiguous along K,
// enabling b128 vector loads in the projection kernel.
// ---------------------------------------------------------------------------
__global__ void gat_cvt_f16(const float* __restrict__ x, const float* __restrict__ W,
                            _Float16* __restrict__ x16, _Float16* __restrict__ W16t) {
    const long long stride = (long long)gridDim.x * blockDim.x;
    long long i = (long long)blockIdx.x * blockDim.x + threadIdx.x;
    for (long long idx = i; idx < (long long)N_NODES * IN_F; idx += stride)
        x16[idx] = (_Float16)x[idx];
    for (long long idx = i; idx < (long long)HEADS * IN_F * OUT_F; idx += stride) {
        const int head = (int)(idx >> 12);           // / (64*64)
        const int rem  = (int)(idx & 4095);
        const int f    = rem >> 6;                   // input feature (K)
        const int o    = rem & 63;                   // output feature (N)
        W16t[((long long)head << 12) + (o << 6) + f] = (_Float16)W[idx];
    }
}

// ---------------------------------------------------------------------------
// Projection: h[n, head, o] = sum_f x[n,f] * W[head,f,o]  via WMMA f16->f32.
// One wave per 16x16 output tile; K=64 -> two v_wmma_f32_16x16x32_f16.
// Operand fetch: per lane, each K-block is two contiguous 8xf16 chunks
// (ISA 16-bit A/B layout) -> four global_load_b128 per operand per wave.
// Tile count: (50000/16) * HEADS * (64/16) = 3125*4*4 = 50000 (exact).
// ---------------------------------------------------------------------------
__global__ void gat_proj_wmma(const _Float16* __restrict__ x16,
                              const _Float16* __restrict__ W16t,
                              float* __restrict__ h) {
    const int wave = threadIdx.x >> 5;
    const int lane = threadIdx.x & 31;
    const long long t = (long long)blockIdx.x * (blockDim.x >> 5) + wave;
    const long long n_tiles = (long long)(N_NODES / 16) * HEADS * (OUT_F / 16);
    if (t >= n_tiles) return;                 // wave-uniform: EXEC stays all-ones

    const int tile_n = (int)(t & 3);          // 16-col slice of OUT_F
    const int head   = (int)((t >> 2) & 3);
    const int tile_m = (int)(t >> 4);         // 16-row slice of nodes
    const int m0 = tile_m * 16;
    const int n0 = tile_n * 16;

    const int idx  = lane & 15;               // A: row M; B: col N
    const int half = lane >> 4;               // K-half selector

    // A: row-major x16, row (m0+idx); B: column-major copy Wt, "row" (n0+idx)
    const _Float16* __restrict__ Ap = x16  + (long long)(m0 + idx) * IN_F;
    const _Float16* __restrict__ Bp = W16t + ((long long)head << 12) + ((n0 + idx) << 6);

    v8f c = {};
    #pragma unroll
    for (int kb = 0; kb < IN_F; kb += 32) {
        // VGPRs 0..3 hold K = kb+half*8+0..7 ; VGPRs 4..7 hold K = kb+16+half*8+0..7
        const v8h a_lo = *(const v8h*)(Ap + kb + half * 8);
        const v8h a_hi = *(const v8h*)(Ap + kb + 16 + half * 8);
        const v8h b_lo = *(const v8h*)(Bp + kb + half * 8);
        const v8h b_hi = *(const v8h*)(Bp + kb + 16 + half * 8);
        const v16h a = __builtin_shufflevector(a_lo, a_hi,
                        0,1,2,3,4,5,6,7,8,9,10,11,12,13,14,15);
        const v16h b = __builtin_shufflevector(b_lo, b_hi,
                        0,1,2,3,4,5,6,7,8,9,10,11,12,13,14,15);
        c = __builtin_amdgcn_wmma_f32_16x16x32_f16(false, a, false, b,
                                                   (short)0, c, false, false);
    }

    // D layout: lane = N + 16*(M>=8); VGPR r -> M = half*8 + r
    float* __restrict__ hp = h + (long long)(m0 + half * 8) * HF + head * OUT_F + n0 + idx;
    #pragma unroll
    for (int r = 0; r < 8; ++r)
        hp[(long long)r * HF] = c[r];
}

// ---------------------------------------------------------------------------
// Per-(node,head) attention halves: s_src/s_dst = dot(h[n,head,:], a_*[head,:])
// ---------------------------------------------------------------------------
__global__ void gat_scores(const float* __restrict__ h,
                           const float* __restrict__ a_src, const float* __restrict__ a_dst,
                           float* __restrict__ s_src, float* __restrict__ s_dst) {
    const int i = blockIdx.x * blockDim.x + threadIdx.x;
    if (i >= N_NODES * HEADS) return;
    const int n  = i >> 2;
    const int hd = i & 3;
    const float4* __restrict__ hp = (const float4*)(h + (long long)n * HF + hd * OUT_F);
    const float4* __restrict__ as = (const float4*)(a_src + hd * OUT_F);
    const float4* __restrict__ ad = (const float4*)(a_dst + hd * OUT_F);
    float ss = 0.0f, sd = 0.0f;
    #pragma unroll
    for (int o = 0; o < OUT_F / 4; ++o) {
        const float4 v = hp[o];
        const float4 s4 = as[o];
        const float4 d4 = ad[o];
        ss += v.x * s4.x + v.y * s4.y + v.z * s4.z + v.w * s4.w;
        sd += v.x * d4.x + v.y * d4.y + v.z * d4.z + v.w * d4.w;
    }
    s_src[i] = ss;
    s_dst[i] = sd;
}

// Sign-split float atomic max -> native global_atomic_max_i32 / min_u32.
// Requires target initialized to -inf.
__device__ inline void atomicMaxF(float* addr, float val) {
    if (val >= 0.0f)
        atomicMax((int*)addr, __float_as_int(val));
    else
        atomicMin((unsigned int*)addr, (unsigned int)__float_as_int(val));
}

__device__ inline float leaky(float v) { return v > 0.0f ? v : ALPHA * v; }

// ---------------------------------------------------------------------------
// Edge pass 1: segment max of LeakyReLU logits into e_max[dst, head].
// ---------------------------------------------------------------------------
__global__ void gat_edge_max(const int* __restrict__ src, const int* __restrict__ dst,
                             const float* __restrict__ s_src, const float* __restrict__ s_dst,
                             float* __restrict__ e_max) {
    const int e = blockIdx.x * blockDim.x + threadIdx.x;
    if (e >= N_EDGES) return;
    const int s = src[e], d = dst[e];
    #pragma unroll
    for (int hd = 0; hd < HEADS; ++hd) {
        const float v = leaky(s_src[s * HEADS + hd] + s_dst[d * HEADS + hd]);
        atomicMaxF(&e_max[d * HEADS + hd], v);
    }
}

// ---------------------------------------------------------------------------
// Edge pass 2: denom[dst,head] += exp(e - e_max[dst,head]).
// ---------------------------------------------------------------------------
__global__ void gat_edge_sum(const int* __restrict__ src, const int* __restrict__ dst,
                             const float* __restrict__ s_src, const float* __restrict__ s_dst,
                             const float* __restrict__ e_max, float* __restrict__ denom) {
    const int e = blockIdx.x * blockDim.x + threadIdx.x;
    if (e >= N_EDGES) return;
    const int s = src[e], d = dst[e];
    #pragma unroll
    for (int hd = 0; hd < HEADS; ++hd) {
        const float v  = leaky(s_src[s * HEADS + hd] + s_dst[d * HEADS + hd]);
        const float ex = __expf(v - e_max[d * HEADS + hd]);
        atomicAdd(&denom[d * HEADS + hd], ex);
    }
}

// ---------------------------------------------------------------------------
// Edge pass 3: out[dst,head,:] += attn * h[src,head,:].
// 64 threads per edge: thread -> (head, 4 consecutive outputs) via float4 gather.
// ---------------------------------------------------------------------------
__global__ void gat_edge_aggregate(const int* __restrict__ src, const int* __restrict__ dst,
                                   const float* __restrict__ s_src, const float* __restrict__ s_dst,
                                   const float* __restrict__ e_max, const float* __restrict__ denom,
                                   const float* __restrict__ h, float* __restrict__ out) {
    const long long gid = (long long)blockIdx.x * blockDim.x + threadIdx.x;
    const long long e = gid >> 6;
    if (e >= N_EDGES) return;
    const int lane = (int)(gid & 63);
    const int hd = lane >> 4;             // head
    const int o4 = (lane & 15) * 4;       // 4 consecutive output features
    const int s = src[e], d = dst[e];

    const float v    = leaky(s_src[s * HEADS + hd] + s_dst[d * HEADS + hd]);
    const float attn = __expf(v - e_max[d * HEADS + hd]) /
                       (denom[d * HEADS + hd] + 1e-16f);

    const float4 hv = *(const float4*)(h + (long long)s * HF + hd * OUT_F + o4);
    float* __restrict__ op = out + (long long)d * HF + hd * OUT_F + o4;
    atomicAdd(op + 0, hv.x * attn);
    atomicAdd(op + 1, hv.y * attn);
    atomicAdd(op + 2, hv.z * attn);
    atomicAdd(op + 3, hv.w * attn);
}

// ---------------------------------------------------------------------------
// Launcher
// ---------------------------------------------------------------------------
extern "C" void kernel_launch(void* const* d_in, const int* in_sizes, int n_in,
                              void* d_out, int out_size, void* d_ws, size_t ws_size,
                              hipStream_t stream) {
    const float* x      = (const float*)d_in[0];   // [N, 64]
    const int*   edges  = (const int*)  d_in[1];   // [2, E]
    const float* W      = (const float*)d_in[2];   // [H, 64, 64]
    const float* a_src  = (const float*)d_in[3];   // [H, 64]
    const float* a_dst  = (const float*)d_in[4];   // [H, 64]
    float*       out    = (float*)d_out;           // [N, 256]

    const int* src = edges;
    const int* dst = edges + N_EDGES;

    // Workspace carve-up (all offsets 256B-aligned)
    char* ws = (char*)d_ws;
    size_t off = 0;
    auto carve = [&](size_t bytes) {
        char* p = ws + off;
        off += (bytes + 255) & ~(size_t)255;
        return p;
    };
    _Float16* x16   = (_Float16*)carve((size_t)N_NODES * IN_F * sizeof(_Float16));       // 6.4 MB
    _Float16* W16t  = (_Float16*)carve((size_t)HEADS * IN_F * OUT_F * sizeof(_Float16)); // 32 KB
    float*    h     = (float*)   carve((size_t)N_NODES * HF * sizeof(float));            // 51.2 MB
    float*    ssrc  = (float*)   carve((size_t)N_NODES * HEADS * sizeof(float));
    float*    sdst  = (float*)   carve((size_t)N_NODES * HEADS * sizeof(float));
    float*    emax  = (float*)   carve((size_t)N_NODES * HEADS * sizeof(float));
    float*    denom = (float*)   carve((size_t)N_NODES * HEADS * sizeof(float));
    (void)ws_size; (void)in_sizes; (void)n_in; (void)out_size;

    // 0) init accumulators
    gat_init<<<4096, 256, 0, stream>>>(out, emax, denom);

    // 1) f32 -> f16 conversion (W transposed per head for contiguous-K B loads)
    gat_cvt_f16<<<4096, 256, 0, stream>>>(x, W, x16, W16t);

    // 2) per-head projection via WMMA (50000 tiles, 4 waves/block)
    gat_proj_wmma<<<(N_NODES / 16) * HEADS * (OUT_F / 16) / 4, 128, 0, stream>>>(x16, W16t, h);

    // 3) node-level attention halves
    gat_scores<<<(N_NODES * HEADS + 255) / 256, 256, 0, stream>>>(h, a_src, a_dst, ssrc, sdst);

    // 4) segment max
    gat_edge_max<<<(N_EDGES + 255) / 256, 256, 0, stream>>>(src, dst, ssrc, sdst, emax);

    // 5) segment exp-sum
    gat_edge_sum<<<(N_EDGES + 255) / 256, 256, 0, stream>>>(src, dst, ssrc, sdst, emax, denom);

    // 6) weighted scatter aggregation (64 threads per edge)
    gat_edge_aggregate<<<(int)(((long long)N_EDGES * 64 + 255) / 256), 256, 0, stream>>>(
        src, dst, ssrc, sdst, emax, denom, h, out);
}
